// CQAttentionLayer_91328184582676
// MI455X (gfx1250) — compile-verified, matrix operand order checked by reading, not run on hardware
//
#include <hip/hip_runtime.h>
#include <hip/hip_bf16.h>

typedef __attribute__((ext_vector_type(16))) _Float16 v16h;
typedef __attribute__((ext_vector_type(8)))  _Float16 v8h;
typedef __attribute__((ext_vector_type(8)))  float    v8f;

static constexpr int CL = 1024;   // context length
static constexpr int QL = 128;    // question length
static constexpr int HD = 128;    // hidden
static constexpr int NB = 64;     // batch

// ---- WMMA fragment loader (CDNA5 16x16x32 f16, wave32) --------------------
// Per-lane data = two contiguous runs of 8 halves at offsets {base, 16+base}
// along the K axis (base = 0 for lanes 0-15, 8 for lanes 16-31).
// A tiles stored row-major [M][K]; B tiles stored column-major [N][K]:
// in both cases "ld" is the stride (in halves) between M-rows / N-columns,
// so one helper serves both. ld must be a multiple of 8.
__device__ __forceinline__ v16h frag_ld(const _Float16* p, int ld, int lane) {
    const int mn   = lane & 15;
    const int base = (lane >> 4) << 3;          // 0 or 8
    const _Float16* r = p + mn * ld + base;
    v8h lo = *(const v8h*)(r);                  // ds_load_b128
    v8h hi = *(const v8h*)(r + 16);             // ds_load_b128
    return __builtin_shufflevector(lo, hi, 0,1,2,3,4,5,6,7,8,9,10,11,12,13,14,15);
}

__device__ __forceinline__ v8f wmma16(v16h a, v16h b, v8f c) {
    return __builtin_amdgcn_wmma_f32_16x16x32_f16(false, a, false, b,
                                                  (short)0, c, false, false);
}

// ---- Kernel 1: S[b,i,j] = cw[i] + qw[j] + sum_h (c*w_cq)[i,h] q[j,h] ------
// grid = NB*16 blocks of 128 threads (4 waves); block: 64 rows x 128 cols
__global__ void k_score(const float* __restrict__ ctx, const float* __restrict__ qst,
                        const float* __restrict__ w, float* __restrict__ S) {
    const int b    = blockIdx.x >> 4;
    const int i0   = (blockIdx.x & 15) << 6;
    const int tid  = threadIdx.x;
    const int lane = tid & 31, wv = tid >> 5;

    __shared__ alignas(16) _Float16 As[64 * 136];   // A: (c*w_cq)[il][h]   [M][K]
    __shared__ alignas(16) _Float16 Bs[128 * 136];  // B: q as [j][h]       [N][K]
    __shared__ float cw_s[64];
    __shared__ float qw_s[128];

    const float* cb   = ctx + (size_t)b * HD * CL;
    const float* qb   = qst + (size_t)b * HD * QL;
    const float* w_q  = w;
    const float* w_c  = w + HD;
    const float* w_cq = w + 2 * HD;

    for (int idx = tid; idx < 64 * 128; idx += 128) {        // coalesced over i
        int h = idx >> 6, il = idx & 63;
        As[il * 136 + h] = (_Float16)(cb[(size_t)h * CL + i0 + il] * w_cq[h]);
    }
    for (int idx = tid; idx < 128 * 128; idx += 128) {       // coalesced over j
        int h = idx >> 7, j = idx & 127;
        Bs[j * 136 + h] = (_Float16)qb[(size_t)h * QL + j];
    }
    if (tid < 64) {
        float s = 0.f;
        for (int h = 0; h < HD; ++h) s += cb[(size_t)h * CL + i0 + tid] * w_c[h];
        cw_s[tid] = s;
    }
    {
        float s = 0.f;
        for (int h = 0; h < HD; ++h) s += qb[(size_t)h * QL + tid] * w_q[h];
        qw_s[tid] = s;
    }
    __syncthreads();

    const int r0 = wv << 4;
    const int mb = (lane >> 4) << 3;
    float* Srow = S + ((size_t)b * CL + (i0 + r0)) * QL;

    v16h afr[4];
#pragma unroll
    for (int ks = 0; ks < 4; ++ks)
        afr[ks] = frag_ld(&As[r0 * 136 + ks * 32], 136, lane);

#pragma unroll
    for (int n = 0; n < 8; ++n) {
        v8f acc = {};
#pragma unroll
        for (int ks = 0; ks < 4; ++ks) {
            v16h bb = frag_ld(&Bs[(n * 16) * 136 + ks * 32], 136, lane);
            acc = wmma16(afr[ks], bb, acc);
        }
        const int j = (n << 4) + (lane & 15);
#pragma unroll
        for (int v = 0; v < 8; ++v) {
            const int M = v + mb;
            Srow[(size_t)M * QL + j] = acc[v] + cw_s[r0 + M] + qw_s[j];
        }
    }
}

// ---- Kernel 2a: partial online softmax stats (4 row-chunks per batch) -----
__global__ void k_colstats(const float* __restrict__ S, float* __restrict__ pmax,
                           float* __restrict__ psum) {
    const int b = blockIdx.x >> 2, part = blockIdx.x & 3, j = threadIdx.x;
    const float* Sb = S + ((size_t)b * CL + part * 256) * QL;
    float m = -1e30f, s = 0.f;
    for (int i = 0; i < 256; ++i) {
        const float x = Sb[(size_t)i * QL + j];
        if (x > m) { s = s * __expf(m - x) + 1.f; m = x; }
        else       { s += __expf(x - m); }
    }
    pmax[(b * 4 + part) * QL + j] = m;
    psum[(b * 4 + part) * QL + j] = s;
}

// ---- Kernel 2b: combine partials ------------------------------------------
__global__ void k_combine(const float* __restrict__ pmax, const float* __restrict__ psum,
                          float* __restrict__ cmax, float* __restrict__ csum) {
    const int b = blockIdx.x, j = threadIdx.x;
    float m = -1e30f;
#pragma unroll
    for (int p = 0; p < 4; ++p) m = fmaxf(m, pmax[(b * 4 + p) * QL + j]);
    float s = 0.f;
#pragma unroll
    for (int p = 0; p < 4; ++p)
        s += psum[(b * 4 + p) * QL + j] * __expf(pmax[(b * 4 + p) * QL + j] - m);
    cmax[b * QL + j] = m;
    csum[b * QL + j] = s;
}

// ---- Kernel 3: s1 = exp(S - max)/sum -> f16 -------------------------------
__global__ void k_norm(const float* __restrict__ S, const float* __restrict__ cmax,
                       const float* __restrict__ csum, _Float16* __restrict__ P) {
    const size_t idx = (size_t)blockIdx.x * blockDim.x + threadIdx.x;
    const int j = (int)(idx & 127);
    const int b = (int)(idx >> 17);                          // CL*QL = 2^17
    const float m   = cmax[b * QL + j];
    const float inv = 1.0f / csum[b * QL + j];
    P[idx] = (_Float16)(__expf(S[idx] - m) * inv);
}

// ---- Kernel 4: T[half] = s1[half]^T @ c[half], two K-halves per batch -----
// grid = NB*2 blocks of 256 threads (8 waves)
__global__ void k_t(const _Float16* __restrict__ P, const float* __restrict__ ctx,
                    float* __restrict__ T) {
    const int b    = blockIdx.x >> 1;
    const int half = blockIdx.x & 1;
    const int tid  = threadIdx.x;
    const int lane = tid & 31, wv = tid >> 5;

    __shared__ alignas(16) _Float16 Apt[128 * 72];  // A: s1^T [j][ik]  [M][K]
    __shared__ alignas(16) _Float16 Bc [128 * 72];  // B: c as [h][ik]  [N][K]

    const _Float16* Pb = P + (size_t)b * CL * QL;
    const float*    cb = ctx + (size_t)b * HD * CL;

    v8f acc[8];
#pragma unroll
    for (int n = 0; n < 8; ++n) acc[n] = (v8f){};

    const int r0 = wv << 4;
    for (int ic0 = half * 512; ic0 < half * 512 + 512; ic0 += 64) {
        __syncthreads();
        for (int idx = tid; idx < 64 * 128; idx += 256) {    // coalesced over j
            int ik = idx >> 7, j = idx & 127;
            Apt[j * 72 + ik] = Pb[(size_t)(ic0 + ik) * QL + j];
        }
        for (int idx = tid; idx < 64 * 128; idx += 256) {    // coalesced over i
            int h = idx >> 6, il = idx & 63;
            Bc[h * 72 + il] = (_Float16)cb[(size_t)h * CL + ic0 + il];
        }
        __syncthreads();
#pragma unroll
        for (int ks = 0; ks < 2; ++ks) {
            v16h a = frag_ld(&Apt[r0 * 72 + ks * 32], 72, lane);
#pragma unroll
            for (int n = 0; n < 8; ++n) {
                v16h bb = frag_ld(&Bc[(n * 16) * 72 + ks * 32], 72, lane);
                acc[n] = wmma16(a, bb, acc[n]);
            }
        }
    }
    float* Tb = T + (size_t)(b * 2 + half) * QL * HD;        // partial slab
    const int mb = (lane >> 4) << 3;
#pragma unroll
    for (int n = 0; n < 8; ++n) {
        const int h = (n << 4) + (lane & 15);
#pragma unroll
        for (int v = 0; v < 8; ++v)
            Tb[(size_t)(r0 + mb + v) * HD + h] = acc[n][v];
    }
}

// ---- Kernel 5: a = s1@q, bb = s1@(T0+T1), write concat output -------------
// grid = NB*16 blocks of 128 threads (4 waves)
__global__ void k_final(const _Float16* __restrict__ P, const float* __restrict__ qst,
                        const float* __restrict__ T, const float* __restrict__ ctx,
                        float* __restrict__ out) {
    const int b  = blockIdx.x >> 4;
    const int i0 = (blockIdx.x & 15) << 6;
    const int tid = threadIdx.x;
    const int lane = tid & 31, wv = tid >> 5;

    __shared__ alignas(16) _Float16 Ps[64 * 136];   // A: s1 [il][j]   [M][K]
    __shared__ alignas(16) _Float16 Qs[128 * 136];  // B1: q [h][j]    [N][K]
    __shared__ alignas(16) _Float16 Ts[128 * 136];  // B2: t [h][j]    [N][K]
    __shared__ float Ra[128 * 68];                  // a  [h][il]
    __shared__ float Rb[128 * 68];                  // bb [h][il]

    const _Float16* Pb = P + (size_t)b * CL * QL;
    const float* qb = qst + (size_t)b * HD * QL;
    const float* T0 = T + (size_t)(b * 2 + 0) * QL * HD;
    const float* T1 = T + (size_t)(b * 2 + 1) * QL * HD;
    const float* cb = ctx + (size_t)b * HD * CL;

    for (int idx = tid; idx < 64 * 128; idx += 128) {
        int il = idx >> 7, j = idx & 127;
        Ps[il * 136 + j] = Pb[(size_t)(i0 + il) * QL + j];
    }
    for (int idx = tid; idx < 128 * 128; idx += 128) {       // straight copy
        int h = idx >> 7, j = idx & 127;
        Qs[h * 136 + j] = (_Float16)qb[(size_t)h * QL + j];
    }
    for (int idx = tid; idx < 128 * 128; idx += 128) {       // coalesced over h
        int j = idx >> 7, h = idx & 127;
        Ts[h * 136 + j] = (_Float16)(T0[(size_t)j * HD + h] + T1[(size_t)j * HD + h]);
    }
    __syncthreads();

    const int r0 = wv << 4;
    const int mb = (lane >> 4) << 3;

    v16h afr[4];
#pragma unroll
    for (int ks = 0; ks < 4; ++ks)
        afr[ks] = frag_ld(&Ps[r0 * 136 + ks * 32], 136, lane);

#pragma unroll
    for (int n = 0; n < 8; ++n) {
        v8f accA = {}, accB = {};
#pragma unroll
        for (int ks = 0; ks < 4; ++ks) {
            v16h b1 = frag_ld(&Qs[(n * 16) * 136 + ks * 32], 136, lane);
            v16h b2 = frag_ld(&Ts[(n * 16) * 136 + ks * 32], 136, lane);
            accA = wmma16(afr[ks], b1, accA);
            accB = wmma16(afr[ks], b2, accB);
        }
        const int h = (n << 4) + (lane & 15);
#pragma unroll
        for (int v = 0; v < 8; ++v) {
            const int il = r0 + mb + v;
            Ra[h * 68 + il] = accA[v];
            Rb[h * 68 + il] = accB[v];
        }
    }
    __syncthreads();

    float* ob = out + (size_t)b * 4 * HD * CL;
    for (int idx = tid; idx < 128 * 64; idx += 128) {
        const int ch = idx >> 6, il = idx & 63;
        const int i  = i0 + il;
        const float cv = cb[(size_t)ch * CL + i];
        const float av = Ra[ch * 68 + il];
        const float bv = Rb[ch * 68 + il];
        ob[(size_t)ch * CL + i]            = cv;
        ob[(size_t)(HD + ch) * CL + i]     = av;
        ob[(size_t)(2 * HD + ch) * CL + i] = cv * av;
        ob[(size_t)(3 * HD + ch) * CL + i] = cv * bv;
    }
}

// ---- launcher --------------------------------------------------------------
extern "C" void kernel_launch(void* const* d_in, const int* in_sizes, int n_in,
                              void* d_out, int out_size, void* d_ws, size_t ws_size,
                              hipStream_t stream) {
    (void)in_sizes; (void)n_in; (void)out_size; (void)ws_size;
    const float* ctx = (const float*)d_in[0];   // (B,H,C)
    const float* qst = (const float*)d_in[1];   // (B,H,Q)
    const float* w   = (const float*)d_in[2];   // (3H,)
    float* out = (float*)d_out;                 // (B,4H,C)

    char* ws = (char*)d_ws;
    const size_t szS  = (size_t)NB * CL * QL * sizeof(float);     // 32 MB
    const size_t szP  = (size_t)NB * CL * QL * sizeof(_Float16);  // 16 MB
    const size_t szV  = (size_t)NB * QL * sizeof(float);          // 32 KB
    const size_t szPV = 4 * szV;                                  // 128 KB
    float*     S    = (float*)ws;
    _Float16*  Pm   = (_Float16*)(ws + szS);
    float*     cmax = (float*)(ws + szS + szP);
    float*     csum = (float*)(ws + szS + szP + szV);
    float*     pmax = (float*)(ws + szS + szP + 2 * szV);
    float*     psum = (float*)(ws + szS + szP + 2 * szV + szPV);
    float*     T    = (float*)(ws + szS + szP + 2 * szV + 2 * szPV);  // 8 MB

    k_score   <<<NB * 16, 128, 0, stream>>>(ctx, qst, w, S);
    k_colstats<<<NB * 4,  128, 0, stream>>>(S, pmax, psum);
    k_combine <<<NB,      128, 0, stream>>>(pmax, psum, cmax, csum);
    k_norm    <<<(NB * CL * QL) / 256, 256, 0, stream>>>(S, cmax, csum, Pm);
    k_t       <<<NB * 2,  256, 0, stream>>>(Pm, ctx, T);
    k_final   <<<NB * 16, 128, 0, stream>>>(Pm, qst, T, ctx, out);
}
